// NMIKDELoss_29686813950247
// MI455X (gfx1250) — compile-verified, matrix operand order checked by reading, not run on hardware
//
#include <hip/hip_runtime.h>
#include <stdint.h>

typedef __attribute__((ext_vector_type(16))) __bf16 v16bf;
typedef __attribute__((ext_vector_type(8)))  float  v8f;

#define N_ROWS 4096
#define DFEAT  64

struct Frag { uint4 a, b; };   // 32 bytes = 16 bf16

static __device__ inline v16bf fragv(const Frag& f) {
    return __builtin_bit_cast(v16bf, f);
}

static __device__ inline v8f wmma_bf16(v16bf a, v16bf b, v8f c) {
    // D = A(16x32 bf16) * B(32x16 bf16) + C(16x16 f32)
    return __builtin_amdgcn_wmma_f32_16x16x32_bf16(false, a, false, b, (short)0, c,
                                                   false, false);
}

// Load one 16x32 bf16 fragment row-slice for this lane.
// ISA 16-bit A layout: lane m (0-15): K = h*8..h*8+7 in VGPR0-3, K = 16+h*8.. in VGPR4-7
// where h = lane/16. Symmetric packing used for the B (column) fragment of the Gram.
static __device__ inline Frag load_frag(const uint16_t* __restrict__ base, int row,
                                        int kb, int hf) {
    const uint16_t* rp = base + (size_t)row * DFEAT + kb * 32 + hf * 8;
    Frag f;
    f.a = *(const uint4*)(rp);
    f.b = *(const uint4*)(rp + 16);
    return f;
}

// ---------------------------------------------------------------------------
// Kernel 1: 4x4 average pool -> 64 features/row, split into bf16 hi + lo,
// and per-row squared norm (f32).
// feat layout: [combo(0..3)][part(hi=0,lo=1)][4096][64] uint16 (bf16 bits)
// sq layout:   [combo][4096] f32      combo = pair*2 + (0=x,1=y)
// ---------------------------------------------------------------------------
__global__ __launch_bounds__(64) void nmi_prep_kernel(
        const float* __restrict__ x1, const float* __restrict__ y1,
        const float* __restrict__ x2, const float* __restrict__ y2,
        uint16_t* __restrict__ feat, float* __restrict__ sq) {
    const int row   = blockIdx.x;
    const int combo = blockIdx.y;           // 0:x1 1:y1 2:x2 3:y2
    const int f     = threadIdx.x;          // 0..63
    const float* in = (combo == 0) ? x1 : (combo == 1) ? y1 : (combo == 2) ? x2 : y2;

    const int c  = f >> 2;
    const int hp = (f >> 1) & 1;
    const int wp = f & 1;
    const float* rp = in + (size_t)row * 1024 + c * 64;   // 16*8*8 per row

    float s = 0.f;
#pragma unroll
    for (int dh = 0; dh < 4; ++dh)
#pragma unroll
        for (int dw = 0; dw < 4; ++dw)
            s += rp[(hp * 4 + dh) * 8 + (wp * 4 + dw)];
    const float val = s * (1.0f / 16.0f);

    // split-compensated bf16: val = hi + lo (+ O(2^-17) rel)
    const uint32_t bits = __float_as_uint(val);
    const float    hif  = __uint_as_float(bits & 0xffff0000u);
    const float    rem  = val - hif;
    const uint16_t hi16 = (uint16_t)(bits >> 16);
    const uint16_t lo16 = (uint16_t)(__float_as_uint(rem) >> 16);

    feat[((size_t)(combo * 2 + 0) * N_ROWS + row) * DFEAT + f] = hi16;
    feat[((size_t)(combo * 2 + 1) * N_ROWS + row) * DFEAT + f] = lo16;

    __shared__ float s64[64];
    s64[f] = val * val;
    __syncthreads();
#pragma unroll
    for (int st = 32; st >= 1; st >>= 1) {
        if (f < st) s64[f] += s64[f + st];
        __syncthreads();
    }
    if (f == 0) sq[(size_t)combo * N_ROWS + row] = s64[0];
}

// ---------------------------------------------------------------------------
// Kernel 2: fused Gram (WMMA bf16 split-compensated) + exp + partial row sums.
// grid = (32 row strips, 8 col chunks, 2 pairs), block = 256 (8 waves).
// Wave w owns row tile rt = strip*8 + w (16 rows); sweeps 32 column tiles.
// x-gram is fully consumed (only kx[8] kept) before y-fragments are loaded,
// halving peak register pressure so A-fragments stay resident (no spills).
// part layout: [pair][s(0=x,1=y,2=xy)][chunk][4096] f32, each slot written once.
// ---------------------------------------------------------------------------
__global__ __launch_bounds__(256, 1) void nmi_gram_kernel(
        const uint16_t* __restrict__ feat,
        const float* __restrict__ sq,
        float* __restrict__ part) {
    const int lane  = threadIdx.x & 31;
    const int wv    = threadIdx.x >> 5;
    const int strip = blockIdx.x;
    const int chunk = blockIdx.y;
    const int pair  = blockIdx.z;

    const int rt   = strip * 8 + wv;        // row tile index (16 rows)
    const int hf   = lane >> 4;             // lane half (K-split per ISA layout)
    const int lrow = lane & 15;             // M (A) / N (B) within tile

    const uint16_t* fxh = feat + (size_t)((pair * 2 + 0) * 2 + 0) * N_ROWS * DFEAT;
    const uint16_t* fxl = feat + (size_t)((pair * 2 + 0) * 2 + 1) * N_ROWS * DFEAT;
    const uint16_t* fyh = feat + (size_t)((pair * 2 + 1) * 2 + 0) * N_ROWS * DFEAT;
    const uint16_t* fyl = feat + (size_t)((pair * 2 + 1) * 2 + 1) * N_ROWS * DFEAT;
    const float* sqx = sq + (size_t)(pair * 2 + 0) * N_ROWS;
    const float* sqy = sq + (size_t)(pair * 2 + 1) * N_ROWS;

    // A fragments (loop invariant): hi & lo, 2 K-blocks, x & y  (64 VGPRs)
    Frag Axh[2], Axl[2], Ayh[2], Ayl[2];
    const int arow = rt * 16 + lrow;
#pragma unroll
    for (int kb = 0; kb < 2; ++kb) {
        Axh[kb] = load_frag(fxh, arow, kb, hf);
        Axl[kb] = load_frag(fxl, arow, kb, hf);
        Ayh[kb] = load_frag(fyh, arow, kb, hf);
        Ayl[kb] = load_frag(fyl, arow, kb, hf);
    }

    // Row squared norms for this lane's 8 C-matrix rows (M = v + 8*hf)
    float sqxi[8], sqyi[8];
#pragma unroll
    for (int v = 0; v < 8; ++v) {
        const int i = rt * 16 + hf * 8 + v;
        sqxi[v] = sqx[i];
        sqyi[v] = sqy[i];
    }

    float accx[8], accy[8], accxy[8];
#pragma unroll
    for (int v = 0; v < 8; ++v) { accx[v] = 0.f; accy[v] = 0.f; accxy[v] = 0.f; }

    const int ct0 = chunk * 32;
    for (int ct = ct0; ct < ct0 + 32; ++ct) {
        const int brow = ct * 16 + lrow;

        // -------- x gram tile --------
        {
            Frag B0h = load_frag(fxh, brow, 0, hf);
            Frag B0l = load_frag(fxl, brow, 0, hf);
            Frag B1h = load_frag(fxh, brow, 1, hf);
            Frag B1l = load_frag(fxl, brow, 1, hf);
            v8f sx = {};
            sx = wmma_bf16(fragv(Axh[0]), fragv(B0h), sx);
            sx = wmma_bf16(fragv(Axh[0]), fragv(B0l), sx);
            sx = wmma_bf16(fragv(Axl[0]), fragv(B0h), sx);
            sx = wmma_bf16(fragv(Axh[1]), fragv(B1h), sx);
            sx = wmma_bf16(fragv(Axh[1]), fragv(B1l), sx);
            sx = wmma_bf16(fragv(Axl[1]), fragv(B1h), sx);

            const float sqxj = sqx[brow];
            float kx[8];
#pragma unroll
            for (int v = 0; v < 8; ++v) {
                const float d2x = fmaxf(sqxi[v] + sqxj - 2.0f * sx[v], 0.0f);
                kx[v] = __expf(-0.5f * d2x);
                accx[v] += kx[v];
            }

            // -------- y gram tile (B registers reused) --------
            Frag C0h = load_frag(fyh, brow, 0, hf);
            Frag C0l = load_frag(fyl, brow, 0, hf);
            Frag C1h = load_frag(fyh, brow, 1, hf);
            Frag C1l = load_frag(fyl, brow, 1, hf);
            v8f sy = {};
            sy = wmma_bf16(fragv(Ayh[0]), fragv(C0h), sy);
            sy = wmma_bf16(fragv(Ayh[0]), fragv(C0l), sy);
            sy = wmma_bf16(fragv(Ayl[0]), fragv(C0h), sy);
            sy = wmma_bf16(fragv(Ayh[1]), fragv(C1h), sy);
            sy = wmma_bf16(fragv(Ayh[1]), fragv(C1l), sy);
            sy = wmma_bf16(fragv(Ayl[1]), fragv(C1h), sy);

            const float sqyj = sqy[brow];
#pragma unroll
            for (int v = 0; v < 8; ++v) {
                const float d2y = fmaxf(sqyi[v] + sqyj - 2.0f * sy[v], 0.0f);
                const float ky  = __expf(-0.5f * d2y);
                accy[v]  += ky;
                accxy[v] += kx[v] * ky;   // exp(-(d2x+d2y)/2) exactly
            }
        }

        // Prefetch next column tile's fragment lines (global_prefetch_b8)
        if (ct + 1 < ct0 + 32) {
            const size_t noff = (size_t)(brow + 16) * DFEAT + hf * 8;
            __builtin_prefetch(fxh + noff, 0, 3);
            __builtin_prefetch(fxl + noff, 0, 3);
            __builtin_prefetch(fyh + noff, 0, 3);
            __builtin_prefetch(fyl + noff, 0, 3);
        }
    }

    // Reduce across the 16 lanes sharing each C row (xor masks stay in-group)
#pragma unroll
    for (int v = 0; v < 8; ++v) {
#pragma unroll
        for (int m = 8; m >= 1; m >>= 1) {
            accx[v]  += __shfl_xor(accx[v],  m, 32);
            accy[v]  += __shfl_xor(accy[v],  m, 32);
            accxy[v] += __shfl_xor(accxy[v], m, 32);
        }
    }

    if (lrow == 0) {   // lanes 0 (rows v) and 16 (rows v+8)
        const size_t base = (size_t)rt * 16 + hf * 8;
        float* px  = part + ((size_t)(pair * 3 + 0) * 8 + chunk) * N_ROWS;
        float* py  = part + ((size_t)(pair * 3 + 1) * 8 + chunk) * N_ROWS;
        float* pxy = part + ((size_t)(pair * 3 + 2) * 8 + chunk) * N_ROWS;
#pragma unroll
        for (int v = 0; v < 8; ++v) {
            px[base + v]  = accx[v];
            py[base + v]  = accy[v];
            pxy[base + v] = accxy[v];
        }
    }
}

// ---------------------------------------------------------------------------
// Kernel 3: deterministic final reduction -> NMI per pair -> mean.
// ---------------------------------------------------------------------------
__global__ __launch_bounds__(256) void nmi_reduce_kernel(
        const float* __restrict__ part, float* __restrict__ out) {
    const int tid = threadIdx.x;   // 256 threads
    float acc[6];
#pragma unroll
    for (int s = 0; s < 6; ++s) acc[s] = 0.f;

    for (int row = tid; row < N_ROWS; row += 256) {
        for (int ps = 0; ps < 6; ++ps) {
            float rs = 0.f;
            for (int ch = 0; ch < 8; ++ch)
                rs += part[((size_t)ps * 8 + ch) * N_ROWS + row];
            acc[ps] += __logf(rs);
        }
    }

    __shared__ float red[6 * 256];
#pragma unroll
    for (int s = 0; s < 6; ++s) red[s * 256 + tid] = acc[s];
    __syncthreads();
    for (int st = 128; st >= 1; st >>= 1) {
        if (tid < st)
#pragma unroll
            for (int s = 0; s < 6; ++s) red[s * 256 + tid] += red[s * 256 + tid + st];
        __syncthreads();
    }

    if (tid == 0) {
        const float LOGN = 8.3177661667193435f;       // log(4096)
        const float HC   = 58.812066125099046f;       // 32 * log(2*pi)
        float total = 0.f;
        for (int pair = 0; pair < 2; ++pair) {
            const float mlx  = red[(pair * 3 + 0) * 256] * (1.0f / N_ROWS);
            const float mly  = red[(pair * 3 + 1) * 256] * (1.0f / N_ROWS);
            const float mlxy = red[(pair * 3 + 2) * 256] * (1.0f / N_ROWS);
            const float hx = LOGN + HC - mlx;
            const float hy = LOGN + HC - mly;
            const float mi = LOGN + (mlxy - mlx - mly);
            float nmi = 2.0f * mi / (hx + hy + 1e-8f);
            nmi = fminf(fmaxf(nmi, 0.0f), 1.0f);
            total += nmi;
        }
        out[0] = 0.5f * total;
    }
}

extern "C" void kernel_launch(void* const* d_in, const int* in_sizes, int n_in,
                              void* d_out, int out_size, void* d_ws, size_t ws_size,
                              hipStream_t stream) {
    const float* x1 = (const float*)d_in[0];
    const float* y1 = (const float*)d_in[1];
    const float* x2 = (const float*)d_in[2];
    const float* y2 = (const float*)d_in[3];

    // workspace: feat (4 MiB u16) | sq (64 KiB f32) | partial rowsums (768 KiB f32)
    uint16_t* feat = (uint16_t*)d_ws;
    float*    sq   = (float*)((char*)d_ws + (size_t)4 * 1024 * 1024);
    float*    part = (float*)((char*)d_ws + (size_t)4 * 1024 * 1024 + 65536);

    nmi_prep_kernel<<<dim3(N_ROWS, 4), 64, 0, stream>>>(x1, y1, x2, y2, feat, sq);
    nmi_gram_kernel<<<dim3(32, 8, 2), 256, 0, stream>>>(feat, sq, part);
    nmi_reduce_kernel<<<1, 256, 0, stream>>>(part, (float*)d_out);
}
